// SimpleMHA_3487513444937
// MI455X (gfx1250) — matrix operand order repaired and checked by reading, not compiled, not run on hardware
//
#include <hip/hip_runtime.h>
#include <stdint.h>

#define S_LEN  4096
#define DMODEL 1024
#define NH     16
#define HD     64

typedef __attribute__((ext_vector_type(16))) __bf16       v16bf;
typedef __attribute__((ext_vector_type(8)))  float        v8f;
typedef __attribute__((ext_vector_type(4)))  unsigned int u32x4;
typedef __attribute__((ext_vector_type(8)))  unsigned int u32x8;

union Frag {
  v16bf bf;
  u32x4 u[2];
};

__device__ __forceinline__ unsigned short f2bf(float f) {
  unsigned int u = __float_as_uint(f);
  u += 0x7FFFu + ((u >> 16) & 1u);          // round-to-nearest-even
  return (unsigned short)(u >> 16);
}

__device__ __forceinline__ v8f v8f_zero() {
  v8f z;
#pragma unroll
  for (int i = 0; i < 8; ++i) z[i] = 0.0f;
  return z;
}

// A-fragment (16x32 bf16, M x K): lane ln holds row M=ln; elems 0-7 = K(8*hi+0..7),
// elems 8-15 = K(16+8*hi+0..7).  Memory row-major along K.
__device__ __forceinline__ Frag load_afrag(const unsigned short* row, int kk, int hi) {
  Frag f;
  f.u[0] = *(const u32x4*)(row + kk + (hi << 3));
  f.u[1] = *(const u32x4*)(row + kk + 16 + (hi << 3));
  return f;
}

// B-fragment (32x16 bf16, K x N): lane ln holds col N=ln; elems 0-7 = K(8*hi+0..7),
// elems 8-15 = K(16+8*hi+0..7).
// Memory is B^T row-major: row n, contiguous along K.
__device__ __forceinline__ Frag load_bfrag(const unsigned short* rowT, int kk, int hi) {
  Frag f;
  f.u[0] = *(const u32x4*)(rowT + kk + (hi << 3));
  f.u[1] = *(const u32x4*)(rowT + kk + 16 + (hi << 3));
  return f;
}

__device__ __forceinline__ v8f wmma_bf16(const Frag& a, const Frag& b, v8f c) {
  return __builtin_amdgcn_wmma_f32_16x16x32_bf16(
      false, a.bf, false, b.bf, (short)0, c, false, false);
}

// TDM: DMA a [16 x DMODEL] bf16 tile (rows m0..m0+15 of a [S?,DMODEL] matrix)
// from global memory into LDS.  D# per CDNA5 ISA.
// Issued by one wave; caller must s_wait_tensorcnt + barrier before LDS reads.
__device__ __forceinline__ void tdm_load_tile16(const unsigned short* gsrc,
                                                unsigned short* lds_dst) {
  unsigned long long ga = (unsigned long long)(uintptr_t)gsrc;
  unsigned int lo = (unsigned int)(uintptr_t)lds_dst;   // low 32 bits = LDS offset
  u32x4 g0;
  g0[0] = 1u;                                           // count=1, user descriptor
  g0[1] = lo;                                           // lds_addr
  g0[2] = (unsigned int)ga;                             // global_addr[31:0]
  g0[3] = (unsigned int)(ga >> 32) | 0x80000000u;       // global_addr[56:32] | type=2
  u32x8 g1;
  g1[0] = 0x00010000u;                                  // data_size=1 (2 bytes)
  g1[1] = (DMODEL & 0xFFFFu) << 16;                     // tensor_dim0[15:0]
  g1[2] = (DMODEL >> 16) | ((S_LEN & 0xFFFFu) << 16);   // dim0[31:16] | tensor_dim1[15:0]
  g1[3] = (S_LEN >> 16) | ((unsigned)DMODEL << 16);     // dim1[31:16] | tile_dim0
  g1[4] = 16u;                                          // tile_dim1=16, tile_dim2=0
  g1[5] = DMODEL;                                       // tensor_dim0_stride[31:0]
  g1[6] = 0u;                                           // stride hi | dim1_stride lo
  g1[7] = 0u;
  asm volatile("tensor_load_to_lds %0, %1" :: "s"(g0), "s"(g1) : "memory");
}

// ---------------------------------------------------------------------------
// Kernel 1: fp32 -> bf16 conversion (vectorized x4)
// ---------------------------------------------------------------------------
__global__ __launch_bounds__(256)
void k_conv_bf16(const float* __restrict__ src, unsigned short* __restrict__ dst, int n4) {
  int i = blockIdx.x * blockDim.x + threadIdx.x;
  if (i < n4) {
    float4 f = ((const float4*)src)[i];
    ushort4 o;
    o.x = f2bf(f.x); o.y = f2bf(f.y); o.z = f2bf(f.z); o.w = f2bf(f.w);
    ((ushort4*)dst)[i] = o;
  }
}

// ---------------------------------------------------------------------------
// Kernel 2: fused QKV projection.  All 8 waves of a block share the same
// 16-row A tile -> one TDM load into LDS per block.  Each wave owns one
// 16x16 output tile for Q, K, V simultaneously (A fragment reused 3x).
// B-fragment global loads are double-buffered across the K loop.
// V stored transposed: vt[d][s].
// ---------------------------------------------------------------------------
__global__ __launch_bounds__(256)
void k_qkv_proj(const unsigned short* __restrict__ xb,
                const unsigned short* __restrict__ wq,
                const unsigned short* __restrict__ wk,
                const unsigned short* __restrict__ wv,
                unsigned short* __restrict__ q,
                unsigned short* __restrict__ k,
                unsigned short* __restrict__ vt) {
  __shared__ unsigned short lds_x[16 * DMODEL];   // 32 KB shared A tile

  const int lane = threadIdx.x & 31;
  const int wid  = threadIdx.x >> 5;
  const int tile = blockIdx.x * 8 + wid;          // 16384 tiles total
  const int m0   = (tile >> 6) << 4;              // same for all 8 waves in block
  const int n0   = (tile & 63) << 4;
  const int ln   = lane & 15;
  const int hi   = lane >> 4;

  if (wid == 0) {
    tdm_load_tile16(xb + (size_t)m0 * DMODEL, lds_x);
    __builtin_amdgcn_s_wait_tensorcnt(0);
  }
  __syncthreads();

  const unsigned short* arow  = lds_x + (size_t)ln * DMODEL;   // ds_load_b128
  const unsigned short* bqrow = wq + (size_t)(n0 + ln) * DMODEL;
  const unsigned short* bkrow = wk + (size_t)(n0 + ln) * DMODEL;
  const unsigned short* bvrow = wv + (size_t)(n0 + ln) * DMODEL;

  v8f cq = v8f_zero(), ck = v8f_zero(), cv = v8f_zero();

  Frag bq = load_bfrag(bqrow, 0, hi);
  Frag bk = load_bfrag(bkrow, 0, hi);
  Frag bv = load_bfrag(bvrow, 0, hi);

  for (int kk = 0; kk < DMODEL; kk += 32) {
    Frag a = load_afrag(arow, kk, hi);
    const int kn = (kk + 32 < DMODEL) ? kk + 32 : 0;   // software pipeline B
    __builtin_prefetch(bqrow + kn + 128, 0, 1);
    Frag nbq = load_bfrag(bqrow, kn, hi);
    Frag nbk = load_bfrag(bkrow, kn, hi);
    Frag nbv = load_bfrag(bvrow, kn, hi);
    cq = wmma_bf16(a, bq, cq);
    ck = wmma_bf16(a, bk, ck);
    cv = wmma_bf16(a, bv, cv);
    bq = nbq; bk = nbk; bv = nbv;
  }

  // C layout: lane holds column n0+ln, rows m0 + 8*hi + r, r=0..7
  const int mloc = hi << 3;
#pragma unroll
  for (int r = 0; r < 8; ++r) {
    size_t idx = (size_t)(m0 + mloc + r) * DMODEL + n0 + ln;
    q[idx] = f2bf(cq[r]);
    k[idx] = f2bf(ck[r]);
  }
  // Transposed V store: 8 consecutive s-positions per lane -> one b128 store
  alignas(16) unsigned short pk[8];
#pragma unroll
  for (int r = 0; r < 8; ++r) pk[r] = f2bf(cv[r]);
  *(u32x4*)(vt + (size_t)(n0 + ln) * S_LEN + m0 + mloc) = *(const u32x4*)pk;
}

// ---------------------------------------------------------------------------
// Kernel 3: causal flash attention.  One wave per (head, 16-query tile).
// Online softmax over 32-key blocks; P restaged via per-wave LDS for the
// C-layout -> A-fragment layout conversion.
// ---------------------------------------------------------------------------
__global__ __launch_bounds__(256)
void k_flash(const unsigned short* __restrict__ q,
             const unsigned short* __restrict__ k,
             const unsigned short* __restrict__ vt,
             unsigned short* __restrict__ z) {
  __shared__ unsigned short lds_p[8][16 * 32];    // 1 KB per wave

  const int lane = threadIdx.x & 31;
  const int wid  = threadIdx.x >> 5;
  const int task = blockIdx.x * 8 + wid;          // 4096 tasks
  const int h    = task >> 8;
  const int q0   = (task & 255) << 4;
  const int ln   = lane & 15;
  const int hi   = lane >> 4;
  const int mloc = hi << 3;

  const unsigned short* qrow = q + (size_t)(q0 + ln) * DMODEL + h * HD;
  Frag aq0 = load_afrag(qrow, 0, hi);             // d = 0..31
  Frag aq1 = load_afrag(qrow, 32, hi);            // d = 32..63

  float mrow[8], lrow[8];
  v8f acc[4];
#pragma unroll
  for (int t = 0; t < 4; ++t) acc[t] = v8f_zero();
#pragma unroll
  for (int r = 0; r < 8; ++r) { mrow[r] = -1e30f; lrow[r] = 0.0f; }

  unsigned short* myp = lds_p[wid];
  const float sc = 0.125f;                        // 1/sqrt(64)

  for (int j0 = 0; j0 <= q0 + 15; j0 += 32) {
    const unsigned short* k0row = k + (size_t)(j0 + ln) * DMODEL + h * HD;
    const unsigned short* k1row = k0row + (size_t)16 * DMODEL;

    Frag b0a = load_bfrag(k0row, 0, hi), b0b = load_bfrag(k0row, 32, hi);
    Frag b1a = load_bfrag(k1row, 0, hi), b1b = load_bfrag(k1row, 32, hi);

    v8f s0 = wmma_bf16(aq1, b0b, wmma_bf16(aq0, b0a, v8f_zero()));
    v8f s1 = wmma_bf16(aq1, b1b, wmma_bf16(aq0, b1a, v8f_zero()));

    if (j0 + 31 > q0) {                           // diagonal block: causal mask
#pragma unroll
      for (int r = 0; r < 8; ++r) {
        int qi = q0 + mloc + r;
        float v0 = s0[r] * sc, v1 = s1[r] * sc;
        s0[r] = (j0 + ln      > qi) ? -1e30f : v0;
        s1[r] = (j0 + 16 + ln > qi) ? -1e30f : v1;
      }
    } else {
#pragma unroll
      for (int r = 0; r < 8; ++r) { s0[r] *= sc; s1[r] *= sc; }
    }

    // row max across 32 keys: elementwise then 16-lane butterfly
    float mx[8];
#pragma unroll
    for (int r = 0; r < 8; ++r) mx[r] = fmaxf(s0[r], s1[r]);
#pragma unroll
    for (int off = 1; off < 16; off <<= 1)
#pragma unroll
      for (int r = 0; r < 8; ++r) mx[r] = fmaxf(mx[r], __shfl_xor(mx[r], off, 16));

    float alpha[8];
#pragma unroll
    for (int r = 0; r < 8; ++r) {
      float mn = fmaxf(mrow[r], mx[r]);
      alpha[r] = __expf(mrow[r] - mn);
      mrow[r]  = mn;
    }

    float p0[8], p1[8], sum[8];
#pragma unroll
    for (int r = 0; r < 8; ++r) {
      p0[r] = __expf(s0[r] - mrow[r]);
      p1[r] = __expf(s1[r] - mrow[r]);
      sum[r] = p0[r] + p1[r];
    }
#pragma unroll
    for (int off = 1; off < 16; off <<= 1)
#pragma unroll
      for (int r = 0; r < 8; ++r) sum[r] += __shfl_xor(sum[r], off, 16);
#pragma unroll
    for (int r = 0; r < 8; ++r) lrow[r] = lrow[r] * alpha[r] + sum[r];

#pragma unroll
    for (int t = 0; t < 4; ++t)
#pragma unroll
      for (int r = 0; r < 8; ++r) acc[t][r] *= alpha[r];

    // stage P (16x32) to LDS in bf16, C layout -> row-major
#pragma unroll
    for (int r = 0; r < 8; ++r) {
      myp[(mloc + r) * 32 + ln]      = f2bf(p0[r]);
      myp[(mloc + r) * 32 + 16 + ln] = f2bf(p1[r]);
    }
    asm volatile("s_wait_dscnt 0" ::: "memory");  // same-wave LDS is in-order

    Frag ap = load_afrag(myp + ln * 32, 0, hi);   // reload as A fragment

    // P (16x32) x V (32x64) : 4 WMMAs against transposed V
#pragma unroll
    for (int t = 0; t < 4; ++t) {
      const unsigned short* vrow = vt + (size_t)(h * HD + t * 16 + ln) * S_LEN + j0;
      Frag bv = load_bfrag(vrow, 0, hi);
      acc[t] = wmma_bf16(ap, bv, acc[t]);
    }
  }

  float inv[8];
#pragma unroll
  for (int r = 0; r < 8; ++r) inv[r] = 1.0f / lrow[r];
#pragma unroll
  for (int t = 0; t < 4; ++t)
#pragma unroll
    for (int r = 0; r < 8; ++r)
      z[(size_t)(q0 + mloc + r) * DMODEL + h * HD + t * 16 + ln] =
          f2bf(acc[t][r] * inv[r]);
}

// ---------------------------------------------------------------------------
// Kernel 4: output projection  out = z @ Wo^T  (fp32 output).
// Same TDM-staged A tile + pipelined B loads as the QKV kernel.
// ---------------------------------------------------------------------------
__global__ __launch_bounds__(256)
void k_oproj(const unsigned short* __restrict__ zb,
             const unsigned short* __restrict__ wo,
             float* __restrict__ out) {
  __shared__ unsigned short lds_z[16 * DMODEL];   // 32 KB shared A tile

  const int lane = threadIdx.x & 31;
  const int wid  = threadIdx.x >> 5;
  const int tile = blockIdx.x * 8 + wid;
  const int m0   = (tile >> 6) << 4;
  const int n0   = (tile & 63) << 4;
  const int ln   = lane & 15;
  const int hi   = lane >> 4;

  if (wid == 0) {
    tdm_load_tile16(zb + (size_t)m0 * DMODEL, lds_z);
    __builtin_amdgcn_s_wait_tensorcnt(0);
  }
  __syncthreads();

  const unsigned short* arow = lds_z + (size_t)ln * DMODEL;
  const unsigned short* brow = wo + (size_t)(n0 + ln) * DMODEL;

  v8f c = v8f_zero();
  Frag b = load_bfrag(brow, 0, hi);
  for (int kk = 0; kk < DMODEL; kk += 32) {
    Frag a = load_afrag(arow, kk, hi);
    const int kn = (kk + 32 < DMODEL) ? kk + 32 : 0;
    __builtin_prefetch(brow + kn + 128, 0, 1);
    Frag nb = load_bfrag(brow, kn, hi);
    c = wmma_bf16(a, b, c);
    b = nb;
  }
  const int mloc = hi << 3;
#pragma unroll
  for (int r = 0; r < 8; ++r)
    out[(size_t)(m0 + mloc + r) * DMODEL + n0 + ln] = c[r];
}

// ---------------------------------------------------------------------------
extern "C" void kernel_launch(void* const* d_in, const int* in_sizes, int n_in,
                              void* d_out, int out_size, void* d_ws, size_t ws_size,
                              hipStream_t stream) {
  const float* x  = (const float*)d_in[0];
  const float* Wq = (const float*)d_in[1];
  const float* Wk = (const float*)d_in[2];
  const float* Wv = (const float*)d_in[3];
  const float* Wo = (const float*)d_in[4];
  float* out = (float*)d_out;

  unsigned short* ws = (unsigned short*)d_ws;
  size_t off = 0;
  unsigned short* xb  = ws + off; off += (size_t)S_LEN * DMODEL;
  unsigned short* wqb = ws + off; off += (size_t)DMODEL * DMODEL;
  unsigned short* wkb = ws + off; off += (size_t)DMODEL * DMODEL;
  unsigned short* wvb = ws + off; off += (size_t)DMODEL * DMODEL;
  unsigned short* wob = ws + off; off += (size_t)DMODEL * DMODEL;
  unsigned short* qb  = ws + off; off += (size_t)S_LEN * DMODEL;
  unsigned short* kb  = ws + off; off += (size_t)S_LEN * DMODEL;
  unsigned short* vtb = ws + off; off += (size_t)S_LEN * DMODEL;  // [DMODEL][S]
  unsigned short* zb  = ws + off; off += (size_t)S_LEN * DMODEL;

  const int n4x = S_LEN * DMODEL / 4;
  const int n4w = DMODEL * DMODEL / 4;
  k_conv_bf16<<<(n4x + 255) / 256, 256, 0, stream>>>(x,  xb,  n4x);
  k_conv_bf16<<<(n4w + 255) / 256, 256, 0, stream>>>(Wq, wqb, n4w);
  k_conv_bf16<<<(n4w + 255) / 256, 256, 0, stream>>>(Wk, wkb, n4w);
  k_conv_bf16<<<(n4w + 255) / 256, 256, 0, stream>>>(Wv, wvb, n4w);
  k_conv_bf16<<<(n4w + 255) / 256, 256, 0, stream>>>(Wo, wob, n4w);

  k_qkv_proj<<<2048, 256, 0, stream>>>(xb, wqb, wkb, wvb, qb, kb, vtb);
  k_flash  <<< 512, 256, 0, stream>>>(qb, kb, vtb, zb);
  k_oproj  <<<2048, 256, 0, stream>>>(zb, wob, out);
}
